// GeoKNO_32719060861317
// MI455X (gfx1250) — matrix operand order repaired and checked by reading, not compiled, not run on hardware
//
#include <hip/hip_runtime.h>
#include <hip/hip_bf16.h>

// GeoKNO pipeline for gfx1250 (MI455X), wave32 + WMMA.
// All large einsums -> v_wmma_f32_16x16x32_f16 (f16 operands, f32 accum).
constexpr int NB  = 4;       // batch
constexpr int NN  = 16384;   // nodes
constexpr int NEg = 131072;  // edges
constexpr int NC  = 64;      // channels C
constexpr int NK  = 128;     // fourier modes K
constexpr int NL  = 4;       // layers
constexpr int NH  = 64;      // mlp hidden
constexpr int NFC = 128;     // head hidden

typedef _Float16 half_t;
typedef _Float16 v16h __attribute__((ext_vector_type(16)));
typedef _Float16 v8h  __attribute__((ext_vector_type(8)));
typedef float    v8f  __attribute__((ext_vector_type(8)));

#define FLAG_ACCUM  1   // C += result
#define FLAG_BIASN  2   // add bias[col]
#define FLAG_GELU   4   // gelu epilogue
#define FLAG_OUTF16 8   // write f16 output
#define FLAG_ATOMIC 16  // atomic f32 add (split-K)
#define FLAG_AF32   32  // A operand is f32, convert on the fly

__device__ __forceinline__ float gelu_f(float x) {
    // jax.nn.gelu default (tanh approximation)
    float x3 = x * x * x;
    return 0.5f * x * (1.0f + tanhf(0.7978845608028654f * (x + 0.044715f * x3)));
}

// ---------------------------------------------------------------------------
// Generic WMMA GEMM, specialized at compile time on FLAGS so the inner loop
// is branch-free. C(MxN) = A(MxK) * B(KxN); B supplied "BT-ready":
// BT is an (N x K) row-major array (row = output column, K contiguous).
// Block = 128 threads = 4 waves; block computes a 64(M) x 16(N) tile.
// Each stage loads a 16x64 B tile -> 2 WMMAs per barrier pair.
// Fragment layouts per CDNA5 ISA 7.12.2:
//   A f16 16x32: lane<16 -> M=lane, k in {0..7,16..23}; lane>=16 -> {8..15,24..31}
//   B f16 32x16: lane<16 -> N=lane, k in {0..7,16..23}; lane>=16 -> N=lane-16, {8..15,24..31}
//   C/D f32:     VGPR r -> M = r + (lane>=16 ? 8 : 0), N = lane&15
// ---------------------------------------------------------------------------
template <int FLAGS>
__global__ void __launch_bounds__(128)
gemm_wmma(const void* __restrict__ Ap, const half_t* __restrict__ BT,
          float* __restrict__ Cf, half_t* __restrict__ C16,
          const float* __restrict__ biasN,
          int M, int Nn, int Kd, int mtiles, int ksplit,
          long ldA, long ldBT, long ldC, long sA, long sBT, long sC)
{
    __shared__ half_t BTs[16][64];  // staged B tile (16 cols x 64 K), 2 KB

    const int bb    = blockIdx.z;
    const int mblk  = blockIdx.x % mtiles;
    const int split = blockIdx.x / mtiles;
    const int ntile = blockIdx.y * 16;
    const int wave  = threadIdx.x >> 5;
    const int lane  = threadIdx.x & 31;
    const int mtile = mblk * 64 + wave * 16;
    const int kper  = Kd / ksplit;       // multiple of 64 at every call site
    const int k0    = split * kper;

    const half_t* A16 = (const half_t*)Ap;
    const float*  A32 = (const float*)Ap;

    const int mrow  = mtile + (lane & 15);
    const int klo   = (lane >> 4) << 3;   // 0 or 8
    const int nfrag = lane & 15;
    const int kbsel = (lane >> 4) << 4;   // 0 or 16

    // staging role (all 128 threads): row nn, 8 contiguous halves (16B)
    const int nn  = threadIdx.x >> 3;
    const int kq8 = (threadIdx.x & 7) << 3;

    v8f acc = {};
    for (int kk = k0; kk < k0 + kper; kk += 64) {
        const half_t* bsrc = BT + (size_t)bb * sBT + (size_t)(ntile + nn) * ldBT + kk + kq8;
        *(uint4*)&BTs[nn][kq8] = *(const uint4*)bsrc;
        __syncthreads();

#pragma unroll
        for (int u = 0; u < 2; u++) {
            v16h a;
            if (FLAGS & FLAG_AF32) {
                const float* ar = A32 + (size_t)bb * sA + (size_t)mrow * ldA + kk + u * 32 + klo;
#pragma unroll
                for (int i = 0; i < 8; i++) { a[i] = (half_t)ar[i]; a[8 + i] = (half_t)ar[16 + i]; }
            } else {
                const half_t* ar = A16 + (size_t)bb * sA + (size_t)mrow * ldA + kk + u * 32 + klo;
                v8h alo = *(const v8h*)ar;
                v8h ahi = *(const v8h*)(ar + 16);
                if (u == 0)
                    __builtin_prefetch((const void*)(ar + 128), 0, 0);  // next A chunk
#pragma unroll
                for (int i = 0; i < 8; i++) { a[i] = alo[i]; a[8 + i] = ahi[i]; }
            }
            v8h blo = *(const v8h*)&BTs[nfrag][u * 32 + klo];
            v8h bhi = *(const v8h*)&BTs[nfrag][u * 32 + klo + 16];
            v16h bfr;
#pragma unroll
            for (int i = 0; i < 8; i++) { bfr[i] = blo[i]; bfr[8 + i] = bhi[i]; }
            acc = __builtin_amdgcn_wmma_f32_16x16x32_f16(false, a, false, bfr,
                                                         (short)0, acc, false, false);
        }
        __syncthreads();
    }

    const int col  = ntile + (lane & 15);
    const int radd = (lane >> 4) << 3;
#pragma unroll
    for (int r = 0; r < 8; r++) {
        size_t ci = (size_t)bb * sC + (size_t)(mtile + r + radd) * ldC + col;
        float v = acc[r];
        if (FLAGS & FLAG_ATOMIC) {
            __hip_atomic_fetch_add(&Cf[ci], v, __ATOMIC_RELAXED, __HIP_MEMORY_SCOPE_AGENT);
            continue;
        }
        if (FLAGS & FLAG_ACCUM)  v += Cf[ci];
        if (FLAGS & FLAG_BIASN)  v += biasN[col];
        if (FLAGS & FLAG_GELU)   v = gelu_f(v);
        if (FLAGS & FLAG_OUTF16) C16[ci] = (half_t)v; else Cf[ci] = v;
    }
}

// ---------------------------------------------------------------------------
// Fourier bases: bcn/bsn (b,n,K) f16 natural (BT-ready for reconstruction),
// wbcT/wbsT (b,K,n) f16 weighted+scaled (BT-ready for projection).
// Scale: fold node_weights*NN into wb* (keeps f16 out of subnormals); the
// middle kernel divides by NN in f32.
// ---------------------------------------------------------------------------
__global__ void __launch_bounds__(256)
bases_kernel(const float* __restrict__ nodes, const float* __restrict__ mask,
             const float* __restrict__ nw, const float* __restrict__ modes,
             half_t* __restrict__ bcn, half_t* __restrict__ bsn,
             half_t* __restrict__ wbcT, half_t* __restrict__ wbsT)
{
    const int bb = blockIdx.y;
    const int node = blockIdx.x * 256 + threadIdx.x;
    const size_t bn = (size_t)bb * NN + node;
    const float p0 = nodes[bn * 2 + 0], p1 = nodes[bn * 2 + 1];
    const float mk = mask[bn];
    const float wsc = mk * nw[bn] * (float)NN;
    const float TWO_PI = 6.283185307179586f;

    for (int kc = 0; kc < NK; kc += 32) {
        alignas(16) half_t tc[32], ts[32];
#pragma unroll
        for (int j = 0; j < 32; j++) {
            int k = kc + j;
            float t = TWO_PI * (p0 * modes[k * 2] + p1 * modes[k * 2 + 1]);
            float cf = cosf(t), sf = sinf(t);
            tc[j] = (half_t)(cf * mk);
            ts[j] = (half_t)(sf * mk);
            wbcT[((size_t)bb * NK + k) * NN + node] = (half_t)(cf * wsc);
            wbsT[((size_t)bb * NK + k) * NN + node] = (half_t)(sf * wsc);
        }
        uint4* dc = (uint4*)&bcn[bn * NK + kc];
        uint4* ds = (uint4*)&bsn[bn * NK + kc];
#pragma unroll
        for (int i = 0; i < 4; i++) { dc[i] = ((const uint4*)tc)[i]; ds[i] = ((const uint4*)ts)[i]; }
    }
}

// lift: h = x @ fc0_w + fc0_b, written channel-major (for GEMM A/B) and
// node-major (for edge gathers).
__global__ void __launch_bounds__(256)
lift_kernel(const float* __restrict__ x, const float* __restrict__ w,
            const float* __restrict__ bvec, half_t* __restrict__ h16,
            half_t* __restrict__ hT16)
{
    const int bb = blockIdx.y;
    const int node = blockIdx.x * 256 + threadIdx.x;
    const size_t bn = (size_t)bb * NN + node;
    const float x0 = x[bn * 3], x1 = x[bn * 3 + 1], x2 = x[bn * 3 + 2];
    alignas(16) half_t hl[NC];
#pragma unroll
    for (int c = 0; c < NC; c++) {
        float v = x0 * w[c] + x1 * w[NC + c] + x2 * w[2 * NC + c] + bvec[c];
        h16[((size_t)bb * NC + c) * NN + node] = (half_t)v;
        hl[c] = (half_t)v;
    }
    uint4* d = (uint4*)&hT16[bn * NC];
#pragma unroll
    for (int i = 0; i < NC / 8; i++) d[i] = ((const uint4*)hl)[i];
}

// x_0[b][c] = sum_n h[c][n] * mask[n] * node_weights[n]   (f32 reduction)
__global__ void __launch_bounds__(256)
x0_kernel(const half_t* __restrict__ h16, const float* __restrict__ mask,
          const float* __restrict__ nw, float* __restrict__ x0)
{
    const int c = blockIdx.x, bb = blockIdx.y;
    __shared__ float red[256];
    float acc = 0.f;
    for (int node = threadIdx.x; node < NN; node += 256) {
        size_t bn = (size_t)bb * NN + node;
        acc += (float)h16[((size_t)bb * NC + c) * NN + node] * mask[bn] * nw[bn];
    }
    red[threadIdx.x] = acc; __syncthreads();
    for (int s = 128; s > 0; s >>= 1) {
        if (threadIdx.x < s) red[threadIdx.x] += red[threadIdx.x + s];
        __syncthreads();
    }
    if (threadIdx.x == 0) x0[bb * NC + c] = red[0];
}

// Per-mode 64x64 spectral mix (f32 VALU; too small for WMMA to matter):
// fa = 2*f_c, fb = -2*f_s written as f16 A-operands for reconstruction.
__global__ void __launch_bounds__(64)
middle_kernel(const float* __restrict__ xc_raw, const float* __restrict__ xs_raw,
              const float* __restrict__ wc, const float* __restrict__ wsw,
              half_t* __restrict__ fa, half_t* __restrict__ fb)
{
    const int k = blockIdx.x, bb = blockIdx.y, o = threadIdx.x;
    const float inv = 1.0f / (float)NN;
    float fc = 0.f, fs = 0.f;
    for (int i = 0; i < NC; i++) {
        float xcv =  xc_raw[((size_t)bb * NC + i) * NK + k] * inv;
        float xsv = -xs_raw[((size_t)bb * NC + i) * NK + k] * inv;  // ref: x_s = -einsum
        float wcv = wc[((size_t)i * NC + o) * NK + k];
        float wsv = wsw[((size_t)i * NC + o) * NK + k];
        fc += xcv * wcv - xsv * wsv;
        fs += xsv * wcv + xcv * wsv;
    }
    fa[((size_t)bb * NC + o) * NK + k] = (half_t)( 2.0f * fc);
    fb[((size_t)bb * NC + o) * NK + k] = (half_t)(-2.0f * fs);
}

__global__ void __launch_bounds__(64)
f0_kernel(const float* __restrict__ x0, const float* __restrict__ w0,
          float* __restrict__ f0)
{
    const int bb = blockIdx.x, o = threadIdx.x;
    float a = 0.f;
    for (int i = 0; i < NC; i++) a += x0[bb * NC + i] * w0[i * NC + o];
    f0[bb * NC + o] = a;
}

// edge scatter: one wave per edge, native f32 atomics into grad (n, 2C)
__global__ void __launch_bounds__(256)
edge_kernel(const half_t* __restrict__ hT16, const int* __restrict__ edges,
            const float* __restrict__ egw, float* __restrict__ grad)
{
    const int bb = blockIdx.y;
    const int e  = blockIdx.x * 8 + (threadIdx.x >> 5);
    const int lane = threadIdx.x & 31;
    const size_t be = (size_t)bb * NEg + e;
    const int src = edges[be * 2 + 0];
    const int dst = edges[be * 2 + 1];
    const float w0 = egw[be * 2 + 0];
    const float w1 = egw[be * 2 + 1];
    const half_t* fsrc = hT16 + ((size_t)bb * NN + src) * NC;
    const half_t* fdst = hT16 + ((size_t)bb * NN + dst) * NC;
    float* g = grad + ((size_t)bb * NN + src) * (2 * NC);
#pragma unroll
    for (int h = 0; h < 2; h++) {
        int c = lane + h * 32;
        float df = (float)fdst[c] - (float)fsrc[c];
        __hip_atomic_fetch_add(&g[c],      w0 * df, __ATOMIC_RELAXED, __HIP_MEMORY_SCOPE_AGENT);
        __hip_atomic_fetch_add(&g[NC + c], w1 * df, __ATOMIC_RELAXED, __HIP_MEMORY_SCOPE_AGENT);
    }
}

// h_new = gelu(x1 + x2 + f0*mask + w_b) + gelu(x3); last layer: raw sum -> x3out
__global__ void __launch_bounds__(256)
combine_kernel(const float* __restrict__ y12, const float* __restrict__ x3,
               const float* __restrict__ f0, const float* __restrict__ wb,
               const float* __restrict__ mask, half_t* __restrict__ h16,
               half_t* __restrict__ hT16, float* __restrict__ x3out, int last)
{
    const int bb = blockIdx.y;
    const int node = blockIdx.x * 256 + threadIdx.x;
    const size_t bn = (size_t)bb * NN + node;
    const float mk = mask[bn];
    alignas(16) half_t hl[NC];
#pragma unroll 4
    for (int c = 0; c < NC; c++) {
        float v = y12[((size_t)bb * NC + c) * NN + node] + f0[bb * NC + c] * mk + wb[c];
        if (last) {
            x3out[bn * NC + c] = v;
        } else {
            float hv = gelu_f(v) + gelu_f(x3[bn * NC + c]);
            h16[((size_t)bb * NC + c) * NN + node] = (half_t)hv;
            hl[c] = (half_t)hv;
        }
    }
    if (!last) {
        uint4* d = (uint4*)&hT16[bn * NC];
#pragma unroll
        for (int i = 0; i < NC / 8; i++) d[i] = ((const uint4*)hl)[i];
    }
}

// head: out = gelu(h @ fc1 + b1) @ fc2 + b2   (f32 VALU, weights uniform)
__global__ void __launch_bounds__(128)
head_kernel(const float* __restrict__ hT, const float* __restrict__ fc1w,
            const float* __restrict__ fc1b, const float* __restrict__ fc2w,
            const float* __restrict__ fc2b, float* __restrict__ out)
{
    const int bb = blockIdx.y;
    const int node = blockIdx.x * 128 + threadIdx.x;
    const float* hh = hT + ((size_t)bb * NN + node) * NC;
    float hv[NC];
#pragma unroll
    for (int c = 0; c < NC; c++) hv[c] = hh[c];
    float acc = 0.f;
    for (int f = 0; f < NFC; f++) {
        float s = fc1b[f];
#pragma unroll
        for (int c = 0; c < NC; c++) s += hv[c] * fc1w[c * NFC + f];
        acc += gelu_f(s) * fc2w[f];
    }
    out[(size_t)bb * NN + node] = acc + fc2b[0];
}

// convert (and optionally transpose) f32 weights -> f16, per layer slice
__global__ void __launch_bounds__(256)
wconv_kernel(const float* __restrict__ in, half_t* __restrict__ outp,
             int R, int Cc, int doT)
{
    const int l = blockIdx.y;
    const int idx = blockIdx.x * 256 + threadIdx.x;
    if (idx >= R * Cc) return;
    const int r = idx / Cc, c = idx % Cc;
    const float* I = in + (size_t)l * R * Cc;
    half_t* O = outp + (size_t)l * R * Cc;
    if (doT) O[(size_t)c * R + r] = (half_t)I[idx];
    else     O[idx]               = (half_t)I[idx];
}

// ---------------------------------------------------------------------------
extern "C" void kernel_launch(void* const* d_in, const int* in_sizes, int n_in,
                              void* d_out, int out_size, void* d_ws, size_t ws_size,
                              hipStream_t stream)
{
    (void)in_sizes; (void)n_in; (void)out_size; (void)ws_size;
    const float* x_in  = (const float*)d_in[0];
    const float* mask  = (const float*)d_in[1];
    const float* nodes = (const float*)d_in[2];
    const float* nwts  = (const float*)d_in[3];
    const float* egw   = (const float*)d_in[4];
    const float* modes = (const float*)d_in[5];
    const float* fc0w  = (const float*)d_in[6];
    const float* fc0b  = (const float*)d_in[7];
    const float* sp_wc = (const float*)d_in[8];
    const float* sp_ws = (const float*)d_in[9];
    const float* sp_w0 = (const float*)d_in[10];
    const float* w_w   = (const float*)d_in[11];
    const float* w_b   = (const float*)d_in[12];
    const float* gw1w  = (const float*)d_in[13];
    const float* gw1b  = (const float*)d_in[14];
    const float* gw2w  = (const float*)d_in[15];
    const float* gw2b  = (const float*)d_in[16];
    const float* gw3w  = (const float*)d_in[17];
    const float* gw3b  = (const float*)d_in[18];
    const float* fc1w  = (const float*)d_in[19];
    const float* fc1b  = (const float*)d_in[20];
    const float* fc2w  = (const float*)d_in[21];
    const float* fc2b  = (const float*)d_in[22];
    const int*   edges = (const int*)d_in[23];
    float* out = (float*)d_out;

    // workspace layout (~168 MB total), 256B aligned slices
    char* base = (char*)d_ws;
    size_t off = 0;
    auto take = [&](size_t bytes) -> void* {
        void* p = base + off;
        off = (off + bytes + 255) & ~(size_t)255;
        return p;
    };
    half_t* bcn  = (half_t*)take((size_t)NB * NN * NK * 2);
    half_t* bsn  = (half_t*)take((size_t)NB * NN * NK * 2);
    half_t* wbcT = (half_t*)take((size_t)NB * NK * NN * 2);
    half_t* wbsT = (half_t*)take((size_t)NB * NK * NN * 2);
    half_t* h16  = (half_t*)take((size_t)NB * NC * NN * 2);
    half_t* hT16 = (half_t*)take((size_t)NB * NN * NC * 2);
    float*  y12  = (float*) take((size_t)NB * NC * NN * 4);
    float*  x3   = (float*) take((size_t)NB * NN * NC * 4);
    float*  grad = (float*) take((size_t)NB * NN * 2 * NC * 4);
    half_t* g1   = (half_t*)take((size_t)NB * NN * NH * 2);
    half_t* g2   = (half_t*)take((size_t)NB * NN * NH * 2);
    float*  xc   = (float*) take((size_t)NB * NC * NK * 4);
    float*  xs   = (float*) take((size_t)NB * NC * NK * 4);
    float*  x0   = (float*) take((size_t)NB * NC * 4);
    half_t* fa   = (half_t*)take((size_t)NB * NC * NK * 2);
    half_t* fb   = (half_t*)take((size_t)NB * NC * NK * 2);
    float*  f0   = (float*) take((size_t)NB * NC * 4);
    half_t* ww16 = (half_t*)take((size_t)NL * NC * NC * 2);
    half_t* gw1T = (half_t*)take((size_t)NL * NH * 2 * NC * 2);
    half_t* gw2T = (half_t*)take((size_t)NL * NH * NH * 2);
    half_t* gw3T = (half_t*)take((size_t)NL * NC * NH * 2);

#define GEMM(FLAGS, A, BTp, Cfp, C16p, bias, M, Nn, Kd, ksplit,                 \
             ldA, ldBT, ldC, sA, sBT, sC)                                       \
    do {                                                                        \
        int mtiles_ = (M) / 64;                                                 \
        dim3 g_(mtiles_ * (ksplit), (Nn) / 16, NB);                             \
        gemm_wmma<FLAGS><<<g_, dim3(128), 0, stream>>>(                         \
            (A), (BTp), (Cfp), (C16p), (bias), (M), (Nn), (Kd), mtiles_,        \
            (ksplit), (ldA), (ldBT), (ldC), (sA), (sBT), (sC));                 \
    } while (0)

    // weight conversions (once)
    wconv_kernel<<<dim3(16, NL), 256, 0, stream>>>(w_w,  ww16, NC, NC, 0);
    wconv_kernel<<<dim3(32, NL), 256, 0, stream>>>(gw1w, gw1T, 2 * NC, NH, 1);
    wconv_kernel<<<dim3(16, NL), 256, 0, stream>>>(gw2w, gw2T, NH, NH, 1);
    wconv_kernel<<<dim3(16, NL), 256, 0, stream>>>(gw3w, gw3T, NH, NC, 1);

    bases_kernel<<<dim3(NN / 256, NB), 256, 0, stream>>>(nodes, mask, nwts, modes,
                                                         bcn, bsn, wbcT, wbsT);
    lift_kernel<<<dim3(NN / 256, NB), 256, 0, stream>>>(x_in, fc0w, fc0b, h16, hT16);

    for (int l = 0; l < NL; l++) {
        // projections x_c, x_s_raw: (64 x 128) = h(64 x 16384) * wb(16384 x 128)
        hipMemsetAsync(xc, 0, (size_t)NB * NC * NK * 4, stream);
        hipMemsetAsync(xs, 0, (size_t)NB * NC * NK * 4, stream);
        GEMM(FLAG_ATOMIC, h16, wbcT, xc, (half_t*)nullptr, (const float*)nullptr,
             NC, NK, NN, 16, NN, NN, NK, (long)NC * NN, (long)NK * NN, (long)NC * NK);
        GEMM(FLAG_ATOMIC, h16, wbsT, xs, (half_t*)nullptr, (const float*)nullptr,
             NC, NK, NN, 16, NN, NN, NK, (long)NC * NN, (long)NK * NN, (long)NC * NK);

        x0_kernel<<<dim3(NC, NB), 256, 0, stream>>>(h16, mask, nwts, x0);
        middle_kernel<<<dim3(NK, NB), 64, 0, stream>>>(
            xc, xs, sp_wc + (size_t)l * NC * NC * NK, sp_ws + (size_t)l * NC * NC * NK,
            fa, fb);
        f0_kernel<<<dim3(NB), 64, 0, stream>>>(x0, sp_w0 + (size_t)l * NC * NC, f0);

        // reconstruction + pointwise mix fused into y12 via accumulate chain
        GEMM(0, fa, bcn, y12, (half_t*)nullptr, (const float*)nullptr,
             NC, NN, NK, 1, NK, NK, NN, (long)NC * NK, (long)NN * NK, (long)NC * NN);
        GEMM(FLAG_ACCUM, fb, bsn, y12, (half_t*)nullptr, (const float*)nullptr,
             NC, NN, NK, 1, NK, NK, NN, (long)NC * NK, (long)NN * NK, (long)NC * NN);
        GEMM(FLAG_ACCUM, ww16 + (size_t)l * NC * NC, hT16, y12, (half_t*)nullptr,
             (const float*)nullptr, NC, NN, NC, 1, NC, NC, NN, 0L,
             (long)NN * NC, (long)NC * NN);

        if (l < NL - 1) {
            // gradient features + node MLP
            hipMemsetAsync(grad, 0, (size_t)NB * NN * 2 * NC * 4, stream);
            edge_kernel<<<dim3(NEg / 8, NB), 256, 0, stream>>>(hT16, edges, egw, grad);
            GEMM(FLAG_AF32 | FLAG_BIASN | FLAG_GELU | FLAG_OUTF16, grad,
                 gw1T + (size_t)l * NH * 2 * NC, (float*)nullptr, g1, gw1b + l * NH,
                 NN, NH, 2 * NC, 1, 2L * NC, 2L * NC, NH,
                 (long)NN * 2 * NC, 0L, (long)NN * NH);
            GEMM(FLAG_BIASN | FLAG_GELU | FLAG_OUTF16, g1,
                 gw2T + (size_t)l * NH * NH, (float*)nullptr, g2, gw2b + l * NH,
                 NN, NH, NH, 1, (long)NH, (long)NH, (long)NH,
                 (long)NN * NH, 0L, (long)NN * NH);
            GEMM(FLAG_BIASN, g2, gw3T + (size_t)l * NC * NH, x3, (half_t*)nullptr,
                 gw3b + l * NC, NN, NC, NH, 1, (long)NH, (long)NH, (long)NC,
                 (long)NN * NH, 0L, (long)NN * NC);
        }
        combine_kernel<<<dim3(NN / 256, NB), 256, 0, stream>>>(
            y12, x3, f0, w_b + l * NC, mask, h16, hT16, x3, l == NL - 1 ? 1 : 0);
    }

    head_kernel<<<dim3(NN / 128, NB), 128, 0, stream>>>(x3, fc1w, fc1b, fc2w, fc2b, out);
#undef GEMM
}
